// minLSTM_50208167690282
// MI455X (gfx1250) — matrix operand order repaired and checked by reading, not compiled, not run on hardware
//
#include <hip/hip_runtime.h>
#include <hip/hip_bf16.h>
#include <math.h>

// Problem constants (match reference)
constexpr int Bsz = 4;
constexpr int TT  = 4096;
constexpr int DD  = 1024;   // K of the GEMMs
constexpr int HH  = 1024;   // N of the GEMMs
constexpr int MM  = Bsz * TT;  // 16384 rows

// GEMM tiling
constexpr int BM = 64;      // block rows
constexpr int BN = 64;      // block cols
constexpr int BK = 32;      // K per step == WMMA K
constexpr int LDST = 40;    // LDS row stride in bf16 (32 + 8 pad -> 80B rows)

// Scan chunking
constexpr int CH = 64;      // chunks per sequence
constexpr int CL = 64;      // chunk length (CH*CL == TT)

typedef __attribute__((ext_vector_type(16))) __bf16 v16bf;
typedef __attribute__((ext_vector_type(8)))  __bf16 v8bf;
typedef __attribute__((ext_vector_type(8)))  float  v8f;

// ---------- math helpers ----------
__device__ __forceinline__ float softplus_f(float x) {
    // max(x,0) + log1p(exp(-|x|)) : stable softplus
    return fmaxf(x, 0.0f) + log1pf(__expf(-fabsf(x)));
}
__device__ __forceinline__ float logaddexp_f(float u, float v) {
    float m = fmaxf(u, v);
    if (m == -INFINITY) return -INFINITY;
    return m + log1pf(__expf(fminf(u, v) - m));
}
__device__ __forceinline__ float log_g_f(float x) {
    // log(g(x)), g(x) = x+0.5 (x>=0) else sigmoid(x)
    return (x >= 0.0f) ? __logf(x + 0.5f) : -softplus_f(-x);
}

// ---------- WMMA fragment helpers ----------
__device__ __forceinline__ v16bf make16(v8bf a, v8bf b) {
    return __builtin_shufflevector(a, b, 0,1,2,3,4,5,6,7,8,9,10,11,12,13,14,15);
}
// A (16x32 bf16, MxK): lane L<16 holds row L, K = 0..7 & 16..23 ; lane L+16: K = 8..15 & 24..31
__device__ __forceinline__ v16bf load_frag_a(const __bf16* base, int lane) {
    int row = lane & 15;
    int kb  = (lane >> 4) * 8;
    const __bf16* p = base + row * LDST + kb;
    v8bf lo = *(const v8bf*)p;
    v8bf hi = *(const v8bf*)(p + 16);
    return make16(lo, hi);
}
// B (32x16 bf16, KxN) fed from row-major W (N rows x K): lane L<16 holds N-row L, K=0..7 & 16..23;
// lane L+16 holds N-row L, K=8..15 & 24..31.
__device__ __forceinline__ v16bf load_frag_b(const __bf16* base, int lane) {
    int row = lane & 15;
    int kb  = (lane >> 4) * 8;
    const __bf16* p = base + row * LDST + kb;
    v8bf lo = *(const v8bf*)p;
    v8bf hi = *(const v8bf*)(p + 16);
    return make16(lo, hi);
}
__device__ __forceinline__ v8f wmma_bf16(v16bf a, v16bf b, v8f c) {
    return __builtin_amdgcn_wmma_f32_16x16x32_bf16(
        /*neg_a=*/false, a, /*neg_b=*/false, b,
        /*c_mod=*/(short)0, c, /*reuse_a=*/false, /*reuse_b=*/false);
}

// =====================================================================
// Kernel 1: fused 3-GEMM (zf,zi,zh) with bf16x3 split-precision WMMA,
// gate log-math fused in epilogue. Writes log_f -> logf_out (d_out scratch)
// and log_v -> logv_out (workspace).
// Block: 256 threads = 8 waves arranged 2(M) x 4(N); wave tile 32x16;
// per wave per K-step: 2 subtiles x 3 gates x 3 passes = 18 v_wmma.
// =====================================================================
__global__ __launch_bounds__(256)
void minlstm_gates_gemm(const float* __restrict__ x,
                        const float* __restrict__ Wf, const float* __restrict__ bf_,
                        const float* __restrict__ Wi, const float* __restrict__ bi_,
                        const float* __restrict__ Wh, const float* __restrict__ bh_,
                        float* __restrict__ logf_out, float* __restrict__ logv_out) {
    __shared__ __bf16 sA[2][BM * LDST];        // [hi/lo][row*LDST+k]
    __shared__ __bf16 sB[3][2][BN * LDST];     // [gate][hi/lo][wrow*LDST+k]

    const int tid   = threadIdx.x;
    const int lane  = tid & 31;
    const int wave  = tid >> 5;
    const int waveM = wave >> 2;    // 0..1
    const int waveN = wave & 3;     // 0..3

    const int row0 = blockIdx.y * BM;   // M base (over B*T)
    const int n0   = blockIdx.x * BN;   // N base (over H)

    const float* Ws[3] = { Wf, Wi, Wh };

    v8f acc[2][3];
    const v8f vzero = {};
    #pragma unroll
    for (int s = 0; s < 2; ++s)
        #pragma unroll
        for (int g = 0; g < 3; ++g) acc[s][g] = vzero;

    for (int k0 = 0; k0 < DD; k0 += BK) {
        __syncthreads();   // previous iteration's LDS reads done

        // ---- stage A tile: 64x32 fp32 -> bf16 hi/lo (512 float4, 2 per thread)
        #pragma unroll
        for (int i = 0; i < 2; ++i) {
            int idx = tid + i * 256;         // 0..511
            int r   = idx >> 3;
            int c4  = idx & 7;
            const float4 v = *(const float4*)(x + (size_t)(row0 + r) * DD + k0 + (c4 << 2));
            float vv[4] = { v.x, v.y, v.z, v.w };
            int off = r * LDST + (c4 << 2);
            #pragma unroll
            for (int j = 0; j < 4; ++j) {
                __bf16 hi = (__bf16)vv[j];
                __bf16 lo = (__bf16)(vv[j] - (float)hi);
                sA[0][off + j] = hi;
                sA[1][off + j] = lo;
            }
        }
        // ---- stage 3 B tiles (W rows n0..n0+63, cols k0..k0+31)
        #pragma unroll
        for (int g = 0; g < 3; ++g) {
            #pragma unroll
            for (int i = 0; i < 2; ++i) {
                int idx = tid + i * 256;
                int r   = idx >> 3;
                int c4  = idx & 7;
                const float4 v = *(const float4*)(Ws[g] + (size_t)(n0 + r) * DD + k0 + (c4 << 2));
                float vv[4] = { v.x, v.y, v.z, v.w };
                int off = r * LDST + (c4 << 2);
                #pragma unroll
                for (int j = 0; j < 4; ++j) {
                    __bf16 hi = (__bf16)vv[j];
                    __bf16 lo = (__bf16)(vv[j] - (float)hi);
                    sB[g][0][off + j] = hi;
                    sB[g][1][off + j] = lo;
                }
            }
        }
        __syncthreads();

        // ---- fragments
        v16bf ahi[2], alo[2];
        #pragma unroll
        for (int s = 0; s < 2; ++s) {
            const int abase = (waveM * 32 + s * 16) * LDST;
            ahi[s] = load_frag_a(&sA[0][abase], lane);
            alo[s] = load_frag_a(&sA[1][abase], lane);
        }
        v16bf bhi[3], blo[3];
        #pragma unroll
        for (int g = 0; g < 3; ++g) {
            const int bbase = (waveN * 16) * LDST;
            bhi[g] = load_frag_b(&sB[g][0][bbase], lane);
            blo[g] = load_frag_b(&sB[g][1][bbase], lane);
        }

        // ---- bf16x3 WMMA: hi*hi + hi*lo + lo*hi, f32 accumulate
        #pragma unroll
        for (int s = 0; s < 2; ++s)
            #pragma unroll
            for (int g = 0; g < 3; ++g) {
                acc[s][g] = wmma_bf16(ahi[s], bhi[g], acc[s][g]);
                acc[s][g] = wmma_bf16(ahi[s], blo[g], acc[s][g]);
                acc[s][g] = wmma_bf16(alo[s], bhi[g], acc[s][g]);
            }
    }

    // ---- epilogue: gate log-math, write log_f / log_v
    const int col = n0 + waveN * 16 + (lane & 15);
    const float bfv = bf_[col], biv = bi_[col], bhv = bh_[col];
    #pragma unroll
    for (int s = 0; s < 2; ++s) {
        // C layout: VGPR j -> M=j (lanes 0..15), M=8+j (lanes 16..31); N = lane&15
        const int rbase = row0 + waveM * 32 + s * 16 + ((lane >> 4) << 3);
        #pragma unroll
        for (int j = 0; j < 8; ++j) {
            const float zf = acc[s][0][j] + bfv;
            const float zi = acc[s][1][j] + biv;
            const float zh = acc[s][2][j] + bhv;
            const float diff  = softplus_f(-zf) - softplus_f(-zi);
            const float log_f = -softplus_f(diff);
            const float log_i = -softplus_f(-diff);
            const float log_v = log_i + log_g_f(zh);
            const size_t o = (size_t)(rbase + j) * HH + col;
            logf_out[o] = log_f;
            logv_out[o] = log_v;
        }
    }
}

// =====================================================================
// Kernel 2a: per-(channel,chunk) reduce of the log-linear recurrence to
// the pair (a = sum log_f, b = chunk-local scan from -inf).
// =====================================================================
__global__ __launch_bounds__(256)
void scan_chunk_reduce(const float* __restrict__ logf, const float* __restrict__ logv,
                       float* __restrict__ abuf, float* __restrict__ bbuf) {
    const int idx   = blockIdx.x * 256 + threadIdx.x;   // over B*CH*H
    const int h     = idx & (HH - 1);
    const int t2    = idx >> 10;
    const int chunk = t2 & (CH - 1);
    const int b     = t2 >> 6;
    size_t base = ((size_t)(b * TT + chunk * CL)) * HH + h;

    float a = 0.0f;
    float bb = -INFINITY;
    for (int t = 0; t < CL; ++t) {
        const float lf = logf[base];
        const float lv = logv[base];
        a += lf;
        bb = logaddexp_f(lf + bb, lv);
        base += HH;
    }
    const size_t o = ((size_t)(b * CH + chunk)) * HH + h;
    abuf[o] = a;
    bbuf[o] = bb;
}

// =====================================================================
// Kernel 2b: per-channel sequential prefix over the CH chunk pairs,
// seeded with log_g(h0). sbuf[c] = log_h at START of chunk c.
// =====================================================================
__global__ __launch_bounds__(256)
void scan_chunk_prefix(const float* __restrict__ h0,
                       const float* __restrict__ abuf, const float* __restrict__ bbuf,
                       float* __restrict__ sbuf) {
    const int idx = blockIdx.x * 256 + threadIdx.x;     // over B*H
    const int h   = idx & (HH - 1);
    const int b   = idx >> 10;
    float s = log_g_f(h0[b * HH + h]);
    for (int c = 0; c < CH; ++c) {
        const size_t o = ((size_t)(b * CH + c)) * HH + h;
        sbuf[o] = s;
        s = logaddexp_f(abuf[o] + s, bbuf[o]);
    }
}

// =====================================================================
// Kernel 2c: replay each chunk from its start state and emit exp(log_h).
// Reads log_f from `out` (element-wise read-before-overwrite) and log_v
// from workspace; overwrites `out` with the final hidden states.
// =====================================================================
__global__ __launch_bounds__(256)
void scan_chunk_apply(const float* __restrict__ logv, const float* __restrict__ sbuf,
                      float* __restrict__ out) {
    const int idx   = blockIdx.x * 256 + threadIdx.x;   // over B*CH*H
    const int h     = idx & (HH - 1);
    const int t2    = idx >> 10;
    const int chunk = t2 & (CH - 1);
    const int b     = t2 >> 6;
    size_t base = ((size_t)(b * TT + chunk * CL)) * HH + h;

    float s = sbuf[((size_t)(b * CH + chunk)) * HH + h];
    for (int t = 0; t < CL; ++t) {
        const float lf = out[base];    // log_f (scratch), read before overwrite
        const float lv = logv[base];
        s = logaddexp_f(lf + s, lv);
        out[base] = __expf(s);
        base += HH;
    }
}

// =====================================================================
extern "C" void kernel_launch(void* const* d_in, const int* in_sizes, int n_in,
                              void* d_out, int out_size, void* d_ws, size_t ws_size,
                              hipStream_t stream) {
    const float* x  = (const float*)d_in[0];
    const float* h0 = (const float*)d_in[1];
    const float* Wf = (const float*)d_in[2];
    const float* bf = (const float*)d_in[3];
    const float* Wi = (const float*)d_in[4];
    const float* bi = (const float*)d_in[5];
    const float* Wh = (const float*)d_in[6];
    const float* bh = (const float*)d_in[7];
    float* out = (float*)d_out;

    // Workspace layout: log_v (64MB) | abuf (1MB) | bbuf (1MB) | sbuf (1MB)
    char* ws = (char*)d_ws;
    const size_t MN   = (size_t)MM * HH;          // 16M elements
    const size_t PAIR = (size_t)Bsz * CH * HH;    // 256K elements
    float* logv = (float*)ws;
    float* abuf = (float*)(ws + MN * sizeof(float));
    float* bbuf = abuf + PAIR;
    float* sbuf = bbuf + PAIR;

    // 1) fused gate GEMM (WMMA bf16x3) -> log_f in d_out, log_v in ws
    dim3 grid1(HH / BN, MM / BM);     // (16, 256)
    minlstm_gates_gemm<<<grid1, 256, 0, stream>>>(x, Wf, bf, Wi, bi, Wh, bh, out, logv);

    // 2) chunked parallel log-space scan
    const int nCT = (Bsz * HH * CH) / 256;   // 1024 blocks
    scan_chunk_reduce<<<nCT, 256, 0, stream>>>(out, logv, abuf, bbuf);
    scan_chunk_prefix<<<(Bsz * HH) / 256, 256, 0, stream>>>(h0, abuf, bbuf, sbuf);
    scan_chunk_apply<<<nCT, 256, 0, stream>>>(logv, sbuf, out);
}
